// CurricularFace_17910013625058
// MI455X (gfx1250) — compile-verified, matrix operand order checked by reading, not run on hardware
//
#include <hip/hip_runtime.h>
#include <hip/hip_bf16.h>
#include <math.h>

// ---------------- problem constants ----------------
#define BDIM 512      // batch
#define DDIM 512      // feature dim (K)
#define CDIM 200000   // classes (N)
#define S_SCALE 64.0f
#define COS_M_C 0.8775825618903728f     // cos(0.5)
#define SIN_M_C 0.4794255386042030f     // sin(0.5)
#define THRESH_C (-0.8775825618903728f) // cos(pi - 0.5)
#define MM_C 0.2397127693021015f        // sin(pi-0.5)*0.5
#define EPS_C 1e-12f

#define BK 32                     // K per chunk = one f16 WMMA K-step
#define BN 64
#define ROWPH 40                  // padded half-row stride (32 data + 8 pad)
#define NCHUNK (DDIM / BK)        // 16
#define TPB 512                   // 16 waves

// ---- CDNA5 async global->LDS copy (no VGPR payload, ASYNCcnt tracked) ----
#if defined(__has_builtin)
#if __has_builtin(__builtin_amdgcn_global_load_async_to_lds_b128) && \
    __has_builtin(__builtin_amdgcn_s_wait_asynccnt)
#define HAVE_ASYNC_LDS 1
#endif
#endif

typedef _Float16 half_t;
typedef _Float16 half4v __attribute__((ext_vector_type(4)));
typedef _Float16 half8v __attribute__((ext_vector_type(8)));
typedef _Float16 v16h __attribute__((ext_vector_type(16)));
typedef float v8f __attribute__((ext_vector_type(8)));

#ifdef HAVE_ASYNC_LDS
// parameter type per hipcc diagnostic: vector_size(16) int, AS1/AS3 pointers
typedef int v4i_g __attribute__((vector_size(4 * sizeof(int))));
typedef __attribute__((address_space(1))) v4i_g* gas_ptr;
typedef __attribute__((address_space(3))) v4i_g* las_ptr;
#endif

__device__ __forceinline__ float wave_sum32(float v) {
#pragma unroll
  for (int off = 16; off > 0; off >>= 1) v += __shfl_xor(v, off, 32);
  return v;
}

// ---- kernel 1: normalize feats -> f16, packed fnPackH[chunk][m][32] ----
__global__ __launch_bounds__(256) void k_fn(const float* __restrict__ feats,
                                            half_t* __restrict__ fnPackH) {
  int wave = (int)((blockIdx.x * blockDim.x + threadIdx.x) >> 5);
  int lane = (int)(threadIdx.x & 31);
  if (wave >= BDIM) return;
  const float* row = feats + (size_t)wave * DDIM;
  float4 v[4];
  float s = 0.0f;
#pragma unroll
  for (int j = 0; j < 4; ++j) {
    v[j] = *(const float4*)(row + lane * 4 + j * 128);
    s += v[j].x * v[j].x + v[j].y * v[j].y + v[j].z * v[j].z + v[j].w * v[j].w;
  }
  s = wave_sum32(s);
  float inv = 1.0f / fmaxf(sqrtf(s), EPS_C);
#pragma unroll
  for (int j = 0; j < 4; ++j) {
    int k0 = lane * 4 + j * 128;  // 4 consecutive k, same 32-wide chunk
    int chunk = k0 >> 5;
    int kl = k0 & 31;             // multiple of 4
    half4v h;
    h[0] = (half_t)(v[j].x * inv);
    h[1] = (half_t)(v[j].y * inv);
    h[2] = (half_t)(v[j].z * inv);
    h[3] = (half_t)(v[j].w * inv);
    *(half4v*)(fnPackH + ((size_t)chunk * BDIM + wave) * BK + kl) = h;
  }
}

// ---- kernel 2: per-row target logit (fused norm of W[label]) ----
__global__ __launch_bounds__(256) void k_tl(const half_t* __restrict__ fnPackH,
                                            const float* __restrict__ W,
                                            const int* __restrict__ labels,
                                            float* __restrict__ tl) {
  int wave = (int)((blockIdx.x * blockDim.x + threadIdx.x) >> 5);
  int lane = (int)(threadIdx.x & 31);
  if (wave >= BDIM) return;
  int c = labels[wave];
  const float* wr = W + (size_t)c * DDIM;
  float dot = 0.0f, sq = 0.0f;
#pragma unroll
  for (int j = 0; j < 16; ++j) {
    int k = lane + 32 * j;
    float wv = wr[k];
    float fv =
        (float)fnPackH[((size_t)(k >> 5) * BDIM + wave) * BK + (k & 31)];
    dot += (float)(half_t)wv * fv;  // match f16 GEMM operand rounding
    sq += wv * wv;
  }
  dot = wave_sum32(dot);
  sq = wave_sum32(sq);
  if (lane == 0) {
    float inv = 1.0f / fmaxf(sqrtf(sq), EPS_C);
    float ct = dot * inv;
    tl[wave] = fminf(fmaxf(ct, -1.0f), 1.0f);
  }
}

// ---- kernel 3: t_new + per-row margin quantities ----
__global__ __launch_bounds__(512) void k_stats(const float* __restrict__ tl,
                                               const float* __restrict__ t_in,
                                               float* __restrict__ ctm,
                                               float* __restrict__ ftl,
                                               float* __restrict__ tnew) {
  __shared__ float red[BDIM];
  int m = (int)threadIdx.x;
  float v = tl[m];
  red[m] = v;
  __syncthreads();
#pragma unroll
  for (int s = 256; s > 0; s >>= 1) {
    if (m < s) red[m] += red[m + s];
    __syncthreads();
  }
  if (m == 0) tnew[0] = (red[0] / (float)BDIM) * 0.01f + 0.99f * t_in[0];
  float sin_t = sqrtf(fmaxf(1.0f - v * v, 0.0f));
  float c = v * COS_M_C - sin_t * SIN_M_C;
  ctm[m] = c;
  ftl[m] = (v > THRESH_C) ? c : (v - MM_C);
}

// ---- kernel 4: f16 WMMA GEMM + fused W-norm + curricular epilogue ----
// block: 512 threads = 16 waves; tile M=512 x N=64, K-chunk=32.
// waves = 4(M) x 4(N); each wave: 8 M-subtiles x 1 N-subtile of 16x16,
// v_wmma_f32_16x16x32_f16 (f32 accumulate).
// A chunk moves global->LDS via GLOBAL_LOAD_ASYNC_TO_LDS_B128 (ASYNCcnt):
// zero VGPR payload -> no spills, transfers issue back-to-back. Only bReg
// (4 VGPRs, the HBM-critical W stream) is pipelined across the WMMA burst.
__global__ __launch_bounds__(TPB)
    __attribute__((amdgpu_waves_per_eu(4))) void k_gemm(
        const half_t* __restrict__ fnPackH, const float* __restrict__ W,
        const int* __restrict__ labels, const float* __restrict__ ctm,
        const float* __restrict__ ftl, const float* __restrict__ tnew,
        float* __restrict__ out) {
  __shared__ __align__(16) half_t As[BDIM * ROWPH];  // 40 KB
  __shared__ __align__(16) half_t Bs[BN * ROWPH];    // 5 KB
  __shared__ float partsq[TPB];
  __shared__ float wninv_s[BN];
  __shared__ float ctm_s[BDIM];
  __shared__ float ftl_s[BDIM];
  __shared__ int lab_s[BDIM];

  const int tid = (int)threadIdx.x;
  const int lane = tid & 31;
  const int w = tid >> 5;
  const int wm = w & 3;   // M group: rows [wm*128, wm*128+128)
  const int wn = w >> 2;  // N group: cols [wn*16, wn*16+16)
  const size_t NbaseB = (size_t)blockIdx.x * BN;

  if (tid < BDIM) {
    ctm_s[tid] = ctm[tid];
    ftl_s[tid] = ftl[tid];
    lab_s[tid] = labels[tid];
  }

  v8f acc[8] = {};
  float sq = 0.0f;

  // B-load mapping: thread t -> column cl, 4 K elements at koff
  const int cl = tid >> 3;              // 0..63
  const int koff = (tid & 7) * 4;       // 0,4,...,28
  const float* wrow = W + (NbaseB + (size_t)cl) * DDIM + koff;

  const int ml = lane & 15;
  const int hs8 = (lane >> 4) * 8;      // f16 operand layout: +8 K for hi half

  // only B is pipelined across compute (4 VGPRs)
  float4 bReg = *(const float4*)wrow;

  for (int kc = 0; kc < NCHUNK; ++kc) {
#ifndef HAVE_ASYNC_LDS
    // sync fallback: issue A loads before the barrier; they die at the stores
    uint4 aReg[4];
    {
      const uint4* aSrc = (const uint4*)(fnPackH + (size_t)kc * BDIM * BK);
#pragma unroll
      for (int i = 0; i < 4; ++i) aReg[i] = aSrc[tid + i * TPB];
    }
#endif
    __syncthreads();  // previous chunk fully consumed
#ifdef HAVE_ASYNC_LDS
    // --- async A copy: global -> LDS directly, no VGPR payload ---
    {
      const half_t* aSrc = fnPackH + (size_t)kc * BDIM * BK;
#pragma unroll
      for (int i = 0; i < 4; ++i) {
        int f = tid + i * TPB;
        __builtin_amdgcn_global_load_async_to_lds_b128(
            (gas_ptr)(aSrc + (size_t)f * 8),
            (las_ptr)&As[(f >> 2) * ROWPH + (f & 3) * 8], 0, 0);
      }
    }
#else
    {
#pragma unroll
      for (int i = 0; i < 4; ++i) {
        int f = tid + i * TPB;
        *(uint4*)&As[(f >> 2) * ROWPH + (f & 3) * 8] = aReg[i];
      }
    }
#endif
    // --- stage B (f32 -> f16) + fused sum-of-squares ---
    {
      half4v bh;
      bh[0] = (half_t)bReg.x;
      bh[1] = (half_t)bReg.y;
      bh[2] = (half_t)bReg.z;
      bh[3] = (half_t)bReg.w;
      *(half4v*)&Bs[cl * ROWPH + koff] = bh;
      sq += bReg.x * bReg.x + bReg.y * bReg.y + bReg.z * bReg.z +
            bReg.w * bReg.w;
    }
#ifdef HAVE_ASYNC_LDS
    __builtin_amdgcn_s_wait_asynccnt(0);  // my async transfers landed in LDS
#endif
    __syncthreads();  // LDS ready (everyone's stores + asyncs visible)
    // --- pipelined W load for next chunk (hides under WMMAs) ---
    if (kc + 1 < NCHUNK) {
      bReg = *(const float4*)(wrow + (size_t)(kc + 1) * BK);
    }
    // --- compute: one K=32 step x 8 M-subtiles ---
    {
      const int n = wn * 16 + ml;
      half8v blo = *(const half8v*)&Bs[n * ROWPH + hs8];
      half8v bhi = *(const half8v*)&Bs[n * ROWPH + hs8 + 16];
      v16h bfrag = __builtin_shufflevector(blo, bhi, 0, 1, 2, 3, 4, 5, 6, 7,
                                           8, 9, 10, 11, 12, 13, 14, 15);
#pragma unroll
      for (int mt = 0; mt < 8; ++mt) {
        int m = wm * 128 + mt * 16 + ml;
        half8v alo = *(const half8v*)&As[m * ROWPH + hs8];
        half8v ahi = *(const half8v*)&As[m * ROWPH + hs8 + 16];
        v16h afrag = __builtin_shufflevector(alo, ahi, 0, 1, 2, 3, 4, 5, 6, 7,
                                             8, 9, 10, 11, 12, 13, 14, 15);
        acc[mt] = __builtin_amdgcn_wmma_f32_16x16x32_f16(
            false, afrag, false, bfrag, (short)0, acc[mt], false, false);
      }
    }
  }

  // reduce per-column sum-of-squares -> 1/||w_c||
  __syncthreads();
  partsq[tid] = sq;
  __syncthreads();
  if (tid < BN) {
    float s8 = 0.0f;
#pragma unroll
    for (int j = 0; j < 8; ++j) s8 += partsq[8 * tid + j];
    wninv_s[tid] = 1.0f / fmaxf(sqrtf(s8), EPS_C);
  }
  __syncthreads();

  const float tn = tnew[0];
  const int mhalf = (lane >> 4) * 8;  // C/D layout: lanes 16-31 hold M=r+8
  const int nl = wn * 16 + ml;
  const float winv = wninv_s[nl];
  const size_t n = NbaseB + (size_t)nl;

#pragma unroll
  for (int mt = 0; mt < 8; ++mt) {
    v8f a = acc[mt];
#pragma unroll
    for (int r = 0; r < 8; ++r) {
      const int m = wm * 128 + mt * 16 + mhalf + r;
      float cosv = a[r] * winv;
      cosv = fminf(fmaxf(cosv, -1.0f), 1.0f);
      float v = (cosv > ctm_s[m]) ? cosv * (tn + cosv) : cosv;
      if (lab_s[m] == (int)n) v = ftl_s[m];
      out[(size_t)m * CDIM + n] = v * S_SCALE;
    }
  }
}

// ---------------- host launcher ----------------
extern "C" void kernel_launch(void* const* d_in, const int* in_sizes, int n_in,
                              void* d_out, int out_size, void* d_ws,
                              size_t ws_size, hipStream_t stream) {
  const float* feats = (const float*)d_in[0];
  const int* labels = (const int*)d_in[1];
  const float* weight = (const float*)d_in[2];
  const float* t_in = (const float*)d_in[3];
  float* out = (float*)d_out;

  char* ws = (char*)d_ws;
  half_t* fnPackH = (half_t*)ws;                       // 512*512 halfs = 512KB
  float* tl = (float*)(ws + (size_t)BDIM * DDIM * 2);  // 512
  float* ctm = tl + BDIM;                              // 512
  float* ftl = ctm + BDIM;                             // 512
  float* tnew = ftl + BDIM;                            // 1

  k_fn<<<(BDIM * 32 + 255) / 256, 256, 0, stream>>>(feats, fnPackH);
  k_tl<<<(BDIM * 32 + 255) / 256, 256, 0, stream>>>(fnPackH, weight, labels,
                                                    tl);
  k_stats<<<1, BDIM, 0, stream>>>(tl, t_in, ctm, ftl, tnew);
  k_gemm<<<CDIM / BN, TPB, 0, stream>>>(fnPackH, weight, labels, ctm, ftl,
                                        tnew, out);
}
